// HeteroClassifier_72499047956817
// MI455X (gfx1250) — matrix operand order repaired and checked by reading, not compiled, not run on hardware
//
#include <hip/hip_runtime.h>
#include <math.h>

// Problem constants (match reference)
#define R_      4
#define TOTAL_  32768
#define NNODE_  1024
#define B_      32
#define DH_     256
#define C_      8
#define H_      8
#define NCLS_   10
#define E_      262144          // 2^18
#define NTILES_ (TOTAL_/16)     // 2048

typedef __attribute__((ext_vector_type(16))) __bf16 v16bf;
typedef __attribute__((ext_vector_type(8)))  float  v8f;

// ---------------------------------------------------------------------------
// Degree kernels: deg arrays start at 1.0 (self-loop), count edges, rsqrt.
// ---------------------------------------------------------------------------
__global__ void k_set1(float* a, float* b) {
  int i = blockIdx.x * 256 + threadIdx.x;        // R_*TOTAL_ threads exactly
  a[i] = 1.f; b[i] = 1.f;
}

__global__ void k_degcount(const int* __restrict__ src, const int* __restrict__ dst,
                           float* deg_out, float* deg_in) {
  int i = blockIdx.x * 256 + threadIdx.x;        // R_*E_ threads exactly
  int r = i >> 18;                               // E_ == 2^18
  unsafeAtomicAdd(&deg_out[r * TOTAL_ + src[i]], 1.f);
  unsafeAtomicAdd(&deg_in [r * TOTAL_ + dst[i]], 1.f);
}

__global__ void k_rsqrt2(float* a, float* b) {
  int i = blockIdx.x * 256 + threadIdx.x;
  a[i] = rsqrtf(a[i]);
  b[i] = rsqrtf(b[i]);
}

// ---------------------------------------------------------------------------
// Edge scatter: agg[dst] += x[src] * rsqrt(deg_out[src]).  32 threads / edge,
// 8 floats / thread, fp32 atomics hit L2 (both x and agg are L2-resident).
// ---------------------------------------------------------------------------
__global__ void k_scatter(const float* __restrict__ x, const int* __restrict__ src,
                          const int* __restrict__ dst, const float* __restrict__ rso,
                          float* __restrict__ agg) {
  int idx = blockIdx.x * 256 + threadIdx.x;      // E_*32 threads exactly
  int e = idx >> 5, t = idx & 31;
  int s = src[e], d = dst[e];
  float w = rso[s];
  const float4* xs = (const float4*)(x + (size_t)s * DH_ + t * 8);
  float4 a = xs[0], b = xs[1];
  float* ap = agg + (size_t)d * DH_ + t * 8;
  unsafeAtomicAdd(ap + 0, a.x * w); unsafeAtomicAdd(ap + 1, a.y * w);
  unsafeAtomicAdd(ap + 2, a.z * w); unsafeAtomicAdd(ap + 3, a.w * w);
  unsafeAtomicAdd(ap + 4, b.x * w); unsafeAtomicAdd(ap + 5, b.y * w);
  unsafeAtomicAdd(ap + 6, b.z * w); unsafeAtomicAdd(ap + 7, b.w * w);
}

// ---------------------------------------------------------------------------
// Pack A into WMMA bf16 16x32 fragment order (ISA 7.12.2), fusing self-loop
// and both degree normalizations: a = (agg + x*rsqrt(deg_out))*rsqrt(deg_in).
// Apack layout: [tile][kchunk][lane] of v16bf (32B per lane fragment).
// ---------------------------------------------------------------------------
__global__ void k_packA(const float* __restrict__ agg, const float* __restrict__ x,
                        const float* __restrict__ rso, const float* __restrict__ rsi,
                        v16bf* __restrict__ Apack) {
  int idx  = blockIdx.x * 256 + threadIdx.x;     // NTILES_*8*32 threads exactly
  int lane = idx & 31;
  int c    = (idx >> 5) & 7;
  int tile = idx >> 8;
  int row  = tile * 16 + (lane & 15);
  int kbase = 32 * c + ((lane >> 4) ? 8 : 0);
  float so = rso[row], si = rsi[row];
  const float* ag = agg + (size_t)row * DH_ + kbase;
  const float* xr = x   + (size_t)row * DH_ + kbase;
  v16bf v;
#pragma unroll
  for (int i = 0; i < 16; ++i) {
    int k = 16 * (i >> 3) + (i & 7);             // {0..7, 16..23}
    v[i] = (__bf16)((ag[k] + xr[k] * so) * si);
  }
  Apack[idx] = v;
}

// ---------------------------------------------------------------------------
// Pack W (f32 [R][256][256], row-major K x N) into WMMA bf16 B fragment order:
// Bpack[r][kchunk][half][n][i] = W[r][32c + 8*half + (i&7) + 16*(i>>3)][n]
// ---------------------------------------------------------------------------
__global__ void k_packB(const float* __restrict__ W, v16bf* __restrict__ Bpack) {
  int idx  = blockIdx.x * 256 + threadIdx.x;     // R_*8*2*256 threads exactly
  int n    = idx & 255;
  int half = (idx >> 8) & 1;
  int c    = (idx >> 9) & 7;
  int r    = idx >> 12;
  const float* w = W + (size_t)r * DH_ * DH_ + (32 * c + 8 * half) * DH_ + n;
  v16bf v;
#pragma unroll
  for (int i = 0; i < 16; ++i) v[i] = (__bf16)(w[((i & 7) + 16 * (i >> 3)) * DH_]);
  Bpack[idx] = v;
}

// ---------------------------------------------------------------------------
// GEMM: out[16x256 strip] (+)= A(16x256) @ W(256x256), bf16 WMMA, f32 acc.
// 128 threads = 4 waves; wave w owns N range [64w, 64w+64): 4 tiles, 32 WMMAs.
// first: init with bias; relu: applied after last relation of layer 1.
// ---------------------------------------------------------------------------
__global__ void __launch_bounds__(128)
k_gemm(const v16bf* __restrict__ Apack, const v16bf* __restrict__ Bpack,
       const float* __restrict__ bias, float* __restrict__ out,
       int first, int relu) {
  const int tile = blockIdx.x;                   // m0 = tile*16
  const int lane = threadIdx.x & 31;
  const int wave = threadIdx.x >> 5;
  const int n0   = wave * 64;
  const int half = lane >> 4;
  const int nl   = lane & 15;

  v8f acc[4] = {v8f{}, v8f{}, v8f{}, v8f{}};

#pragma unroll
  for (int c = 0; c < 8; ++c) {
    v16bf a = Apack[(tile * 8 + c) * 32 + lane];
    const v16bf* brow = Bpack + (c * 2 + half) * 256;
#pragma unroll
    for (int t = 0; t < 4; ++t) {
      v16bf b = brow[n0 + t * 16 + nl];
      acc[t] = __builtin_amdgcn_wmma_f32_16x16x32_bf16(
          false, a, false, b, (short)0, acc[t], false, false);
    }
  }

#pragma unroll
  for (int t = 0; t < 4; ++t) {
#pragma unroll
    for (int v = 0; v < 8; ++v) {
      int row = tile * 16 + v + 8 * half;        // C/D layout (ISA 7.12.2)
      int col = n0 + t * 16 + nl;
      float val = acc[t][v] + (first ? bias[col] : out[(size_t)row * DH_ + col]);
      if (relu) val = fmaxf(val, 0.f);
      out[(size_t)row * DH_ + col] = val;
    }
  }
}

// ---------------------------------------------------------------------------
// Cluster-wise masked mean pool + per-cluster linear: pooled[b][key][:]
// ---------------------------------------------------------------------------
__global__ void k_pool(const float* __restrict__ h2, const int* __restrict__ node_label,
                       const int* __restrict__ feat_label, const float* __restrict__ Wr,
                       const float* __restrict__ br, float* __restrict__ pooled) {
  int b = blockIdx.x >> 3;
  int key = blockIdx.x & 7;
  int d = threadIdx.x;                           // 256
  float s = 0.f, cnt = 0.f;
  for (int n = 0; n < NNODE_; ++n) {
    if (node_label[n] == key) {                  // first graph's labels only
      cnt += 1.f;
      s += h2[((size_t)b * NNODE_ + n) * DH_ + d];
    }
  }
  float v = s / fmaxf(cnt, 1.f);
  v *= (feat_label[d] == key) ? 1.f : 0.f;
  __shared__ float sv[DH_];
  sv[d] = v;
  __syncthreads();
  const float* w = Wr + (size_t)key * DH_ * DH_ + (size_t)d * DH_;  // Wr[key].T row
  float o = br[key * DH_ + d];
  for (int kk = 0; kk < DH_; ++kk) o += sv[kk] * w[kk];
  pooled[((size_t)b * C_ + key) * DH_ + d] = o;
}

// Y[i][j] = sum_d X[i][d] * W[d][j]   (256x256 @ 256x256, tiny)
__global__ void k_mat256(const float* __restrict__ X, const float* __restrict__ W,
                         float* __restrict__ Y) {
  int i = blockIdx.x, j = threadIdx.x;
  const float* x = X + (size_t)i * DH_;
  float s = 0.f;
  for (int d = 0; d < DH_; ++d) s += x[d] * W[(size_t)d * DH_ + j];
  Y[(size_t)i * DH_ + j] = s;
}

// Per (head,batch) attention over C=8 clusters.
__global__ void k_attn(const float* __restrict__ q, const float* __restrict__ k,
                       const float* __restrict__ v, float* __restrict__ o) {
  int b = blockIdx.x & 31;
  int h = blockIdx.x >> 5;
  int tid = threadIdx.x;                         // 256
  __shared__ float sc[C_][C_];
  if (tid < 64) {
    int cq = tid >> 3, ck = tid & 7;
    const float* qp = q + ((size_t)b * C_ + cq) * DH_ + h * 32;
    const float* kp = k + ((size_t)b * C_ + ck) * DH_ + h * 32;
    float s = 0.f;
    for (int t = 0; t < 32; ++t) s += qp[t] * kp[t];
    sc[cq][ck] = s * 0.35355339059327373f;       // 1/sqrt(C)
  }
  __syncthreads();
  if (tid < 8) {
    float m = sc[tid][0];
    for (int ck = 1; ck < 8; ++ck) m = fmaxf(m, sc[tid][ck]);
    float e[8], sum = 0.f;
    for (int ck = 0; ck < 8; ++ck) { e[ck] = expf(sc[tid][ck] - m); sum += e[ck]; }
    for (int ck = 0; ck < 8; ++ck) sc[tid][ck] = e[ck] / sum;
  }
  __syncthreads();
  int cq = tid >> 5, t = tid & 31;
  float s = 0.f;
  for (int ck = 0; ck < 8; ++ck)
    s += sc[cq][ck] * v[((size_t)b * C_ + ck) * DH_ + h * 32 + t];
  o[((size_t)b * C_ + cq) * DH_ + h * 32 + t] = s;
}

// ff = relu(o @ lin_w.T + lin_b) + o + x  (both residuals fused)
__global__ void k_ffadd(const float* __restrict__ o, const float* __restrict__ lin_w,
                        const float* __restrict__ lin_b, const float* __restrict__ x,
                        float* __restrict__ ff) {
  int i = blockIdx.x, j = threadIdx.x;
  const float* orow = o + (size_t)i * DH_;
  const float* w = lin_w + (size_t)j * DH_;
  float s = lin_b[j];
  for (int d = 0; d < DH_; ++d) s += orow[d] * w[d];
  s = fmaxf(s, 0.f) + orow[j] + x[(size_t)i * DH_ + j];
  ff[(size_t)i * DH_ + j] = s;
}

// graph_emb[b][d] = mean_c ff[b][c][d]
__global__ void k_graphemb(const float* __restrict__ ff, float* __restrict__ ge) {
  int i = blockIdx.x * 256 + threadIdx.x;        // B_*DH_ threads
  int b = i >> 8, d = i & 255;
  float s = 0.f;
  for (int c = 0; c < C_; ++c) s += ff[((size_t)b * C_ + c) * DH_ + d];
  ge[i] = s * (1.f / C_);
}

// hg_raw[b][d] = mean over 1024 nodes
__global__ void k_hgraw(const float* __restrict__ h2, float* __restrict__ hgr) {
  int i = blockIdx.x * 256 + threadIdx.x;        // B_*DH_ threads
  int b = i >> 8, d = i & 255;
  float s = 0.f;
  for (int n = 0; n < NNODE_; ++n) s += h2[((size_t)b * NNODE_ + n) * DH_ + d];
  hgr[i] = s * (1.f / NNODE_);
}

__global__ void k_recon(const float* __restrict__ ge, const float* __restrict__ hgr,
                        float* __restrict__ out) {
  __shared__ float sm[256];
  int tid = threadIdx.x;
  float s = 0.f;
  for (int i = tid; i < B_ * DH_; i += 256) {
    float d = ge[i] - hgr[i] + 1e-6f;
    s += d * d;
  }
  sm[tid] = s; __syncthreads();
  for (int st = 128; st > 0; st >>= 1) {
    if (tid < st) sm[tid] += sm[tid + st];
    __syncthreads();
  }
  if (tid == 0) out[B_ * NCLS_] = sqrtf(sm[0]);
}

__global__ void k_bn(const float* __restrict__ ge, const float* __restrict__ gamma,
                     const float* __restrict__ beta, float* __restrict__ hg) {
  int d = threadIdx.x;                           // 256
  float mu = 0.f;
  for (int b = 0; b < B_; ++b) mu += ge[b * DH_ + d];
  mu *= (1.f / B_);
  float var = 0.f;
  for (int b = 0; b < B_; ++b) { float x = ge[b * DH_ + d] - mu; var += x * x; }
  var *= (1.f / B_);
  float rs = rsqrtf(var + 1e-5f);
  for (int b = 0; b < B_; ++b)
    hg[b * DH_ + d] = (ge[b * DH_ + d] - mu) * rs * gamma[d] + beta[d];
}

__global__ void k_logits(const float* __restrict__ hg, const float* __restrict__ cls_w,
                         const float* __restrict__ cls_b, float* __restrict__ out) {
  int i = blockIdx.x * 256 + threadIdx.x;
  if (i >= B_ * NCLS_) return;
  int b = i / NCLS_, j = i % NCLS_;
  float s = cls_b[j];
  for (int d = 0; d < DH_; ++d) s += hg[b * DH_ + d] * cls_w[j * DH_ + d];
  out[i] = s;
}

// ---------------------------------------------------------------------------
extern "C" void kernel_launch(void* const* d_in, const int* in_sizes, int n_in,
                              void* d_out, int out_size, void* d_ws, size_t ws_size,
                              hipStream_t stream) {
  (void)in_sizes; (void)n_in; (void)out_size; (void)ws_size;

  const float* h          = (const float*)d_in[0];
  const int*   src        = (const int*)  d_in[1];
  const int*   dst        = (const int*)  d_in[2];
  const int*   node_label = (const int*)  d_in[3];
  const int*   feat_label = (const int*)  d_in[4];
  const float* W1         = (const float*)d_in[5];
  const float* b1         = (const float*)d_in[6];
  const float* W2         = (const float*)d_in[7];
  const float* b2         = (const float*)d_in[8];
  const float* Wr         = (const float*)d_in[9];
  const float* br         = (const float*)d_in[10];
  const float* Wq         = (const float*)d_in[11];
  const float* Wk         = (const float*)d_in[12];
  const float* Wv         = (const float*)d_in[13];
  const float* lin_w      = (const float*)d_in[14];
  const float* lin_b      = (const float*)d_in[15];
  const float* bn_gamma   = (const float*)d_in[16];
  const float* bn_beta    = (const float*)d_in[17];
  const float* cls_w      = (const float*)d_in[18];
  const float* cls_b      = (const float*)d_in[19];
  float* out = (float*)d_out;

  // ---- workspace carve ----
  char* p = (char*)d_ws;
  auto carve = [&](size_t bytes) {
    char* r = p;
    p += (bytes + 255) & ~(size_t)255;
    return (void*)r;
  };
  float* rs_out = (float*)carve(R_ * TOTAL_ * 4);
  float* rs_in  = (float*)carve(R_ * TOTAL_ * 4);
  float* agg    = (float*)carve((size_t)TOTAL_ * DH_ * 4);
  float* h1     = (float*)carve((size_t)TOTAL_ * DH_ * 4);
  float* h2     = (float*)carve((size_t)TOTAL_ * DH_ * 4);
  v16bf* Apack  = (v16bf*)carve((size_t)TOTAL_ * DH_ * 2);          // bf16
  v16bf* Bpack  = (v16bf*)carve((size_t)2 * R_ * 8 * 2 * 256 * 32); // 2 layers
  float* pooled = (float*)carve((size_t)B_ * C_ * DH_ * 4);
  float* qb     = (float*)carve((size_t)B_ * C_ * DH_ * 4);
  float* kb     = (float*)carve((size_t)B_ * C_ * DH_ * 4);
  float* vb     = (float*)carve((size_t)B_ * C_ * DH_ * 4);
  float* ob     = (float*)carve((size_t)B_ * C_ * DH_ * 4);
  float* ffb    = (float*)carve((size_t)B_ * C_ * DH_ * 4);
  float* ge     = (float*)carve((size_t)B_ * DH_ * 4);
  float* hgr    = (float*)carve((size_t)B_ * DH_ * 4);
  float* hg     = (float*)carve((size_t)B_ * DH_ * 4);

  // ---- degrees (self-loop => init 1) ----
  k_set1    <<<R_ * TOTAL_ / 256, 256, 0, stream>>>(rs_out, rs_in);
  k_degcount<<<R_ * E_ / 256,     256, 0, stream>>>(src, dst, rs_out, rs_in);
  k_rsqrt2  <<<R_ * TOTAL_ / 256, 256, 0, stream>>>(rs_out, rs_in);

  // ---- pack weights to bf16 WMMA fragments ----
  const int bpack_rel = 8 * 2 * 256;             // v16bf per relation
  k_packB<<<R_ * bpack_rel / 256, 256, 0, stream>>>(W1, Bpack);
  k_packB<<<R_ * bpack_rel / 256, 256, 0, stream>>>(W2, Bpack + R_ * bpack_rel);

  // ---- two R-GCN layers ----
  for (int layer = 0; layer < 2; ++layer) {
    const float* xin  = layer ? h1 : h;
    float*       xout = layer ? h2 : h1;
    const float* bias = layer ? b2 : b1;
    const v16bf* Bp   = Bpack + layer * R_ * bpack_rel;
    for (int r = 0; r < R_; ++r) {
      hipMemsetAsync(agg, 0, (size_t)TOTAL_ * DH_ * 4, stream);
      k_scatter<<<E_ * 32 / 256, 256, 0, stream>>>(
          xin, src + (size_t)r * E_, dst + (size_t)r * E_, rs_out + r * TOTAL_, agg);
      k_packA<<<NTILES_ * 8 * 32 / 256, 256, 0, stream>>>(
          agg, xin, rs_out + r * TOTAL_, rs_in + r * TOTAL_, Apack);
      k_gemm<<<NTILES_, 128, 0, stream>>>(
          Apack, Bp + r * bpack_rel, bias + r * DH_, xout,
          /*first=*/(r == 0), /*relu=*/(layer == 0 && r == R_ - 1));
    }
  }

  // ---- pooling, attention, heads ----
  k_pool<<<B_ * C_, DH_, 0, stream>>>(h2, node_label, feat_label, Wr, br, pooled);
  k_mat256<<<B_ * C_, DH_, 0, stream>>>(pooled, Wq, qb);
  k_mat256<<<B_ * C_, DH_, 0, stream>>>(pooled, Wk, kb);
  k_mat256<<<B_ * C_, DH_, 0, stream>>>(pooled, Wv, vb);
  k_attn<<<H_ * B_, 256, 0, stream>>>(qb, kb, vb, ob);
  k_ffadd<<<B_ * C_, DH_, 0, stream>>>(ob, lin_w, lin_b, pooled, ffb);
  k_graphemb<<<B_ * DH_ / 256, 256, 0, stream>>>(ffb, ge);
  k_hgraw   <<<B_ * DH_ / 256, 256, 0, stream>>>(h2, hgr);
  k_recon<<<1, 256, 0, stream>>>(ge, hgr, out);
  k_bn<<<1, DH_, 0, stream>>>(ge, bn_gamma, bn_beta, hg);
  k_logits<<<(B_ * NCLS_ + 255) / 256, 256, 0, stream>>>(hg, cls_w, cls_b, out);
}
